// FigLinearFull_29094108463114
// MI455X (gfx1250) — matrix operand order repaired and checked by reading, not compiled, not run on hardware
//
#include <hip/hip_runtime.h>

// Problem constants (match the reference)
#define IN_DIM   4096
#define OUT_DIM  4096
#define M_TOTAL  8192                       // 4 * 2048
#define GROUP_SZ 128
#define NPACKED  (OUT_DIM * IN_DIM / 2)     // 8,388,608

// GEMM tiling: block = 256 threads = 8 wave32, tile 128(M) x 128(N), K step 32.
// Waves arranged 4(M) x 2(N); each wave owns 32x64 = 2x4 wmma accumulators.
#define TILE_M   128
#define TILE_N   128
#define TILE_K   32
#define KSTEPS   (IN_DIM / TILE_K)
#define LDSS     40                          // LDS row stride (elems); 80B rows, 16B-aligned

typedef __attribute__((ext_vector_type(16))) __bf16 v16bf;
typedef __attribute__((ext_vector_type(8)))  __bf16 v8bf;
typedef __attribute__((ext_vector_type(8)))  float  v8f;

// ---------------------------------------------------------------------------
// Kernel 1: dequantize int4 groups + add delta  ->  bf16 W_eff [OUT_DIM][IN_DIM]
// flat[2i] = low nibble, flat[2i+1] = high nibble; group = flat / 128
// ---------------------------------------------------------------------------
__global__ __launch_bounds__(256)
void figlin_dequant_bf16(const int* __restrict__ q_packed,
                         const float* __restrict__ q_scales,
                         const float* __restrict__ q_zeros,
                         const float* __restrict__ delta,
                         __bf16* __restrict__ w_bf) {
    const int i = blockIdx.x * blockDim.x + threadIdx.x;   // packed byte index
    if (i >= NPACKED) return;
    const int q   = q_packed[i];
    const int lo  = q & 0xF;
    const int hi  = (q >> 4) & 0xF;
    const size_t f0 = (size_t)2 * i;
    const size_t f1 = f0 + 1;
    const int g   = i >> 6;                // (2i)/128 ; both nibbles in same group
    const float s = q_scales[g];
    const float z = q_zeros[g];
    const float w0 = ((float)lo - z) * s + delta[f0];
    const float w1 = ((float)hi - z) * s + delta[f1];
    w_bf[f0] = (__bf16)w0;
    w_bf[f1] = (__bf16)w1;
}

// ---------------------------------------------------------------------------
// Kernel 2: C[M_TOTAL, OUT_DIM] = x(bf16) @ W_eff^T + bias   via v_wmma bf16
//   Double-buffered LDS; W tile staged with GLOBAL_LOAD_ASYNC_TO_LDS_B128
//   (ASYNCcnt), X tile converted f32->bf16 in flight through VGPRs.
// ---------------------------------------------------------------------------
__global__ __launch_bounds__(256)
void figlin_wmma_gemm(const float* __restrict__ x,      // [M_TOTAL][IN_DIM] f32
                      const __bf16* __restrict__ w_bf,  // [OUT_DIM][IN_DIM] bf16
                      const float* __restrict__ bias,   // [OUT_DIM]
                      float* __restrict__ out) {        // [M_TOTAL][OUT_DIM]
    __shared__ __bf16 lX[2][TILE_M * LDSS];   // 2 x 10240 B
    __shared__ __bf16 lW[2][TILE_N * LDSS];   // 2 x 10240 B

    const int t    = threadIdx.x;
    const int lane = t & 31;
    const int wave = t >> 5;
    const int half = lane >> 4;     // WMMA lane-half (VGPR layout)
    const int lr   = lane & 15;

    const int nTile = blockIdx.x * TILE_N;
    const int mTile = blockIdx.y * TILE_M;

    const int waveM = (wave & 3) * 32;     // 4 waves along M, 32 rows each
    const int waveN = (wave >> 2) * 64;    // 2 waves along N, 64 cols each

    // Global->LDS staging assignments
    const int xr = t >> 3;                 // X: 8 threads * float4 per 32-col row
    const int xc = (t & 7) * 4;
    const int wr = t >> 2;                 // W: 4 threads * 8 bf16 (16B) per row
    const int wc = (t & 3) * 8;

    // Stage the X tile for K-offset k0 into buffer `buf` (f32 -> bf16 convert)
    auto stageX = [&](int buf, int k0) {
        #pragma unroll
        for (int p = 0; p < 4; ++p) {
            const int row = xr + p * 32;
            const float4 v =
                *(const float4*)(x + (size_t)(mTile + row) * IN_DIM + k0 + xc);
            __bf16* d = &lX[buf][row * LDSS + xc];
            d[0] = (__bf16)v.x; d[1] = (__bf16)v.y;
            d[2] = (__bf16)v.z; d[3] = (__bf16)v.w;
        }
    };
    // Stage the W tile with async global->LDS b128 copies (ASYNCcnt-tracked).
    auto stageW = [&](int buf, int k0) {
        #pragma unroll
        for (int p = 0; p < 2; ++p) {
            const int row = wr + p * 64;
            const unsigned voff =
                ((unsigned)(nTile + row) * (unsigned)IN_DIM +
                 (unsigned)(k0 + wc)) * 2u;                       // byte offset
            const unsigned ldsa =
                (unsigned)(size_t)&lW[buf][row * LDSS + wc];      // LDS byte addr
            asm volatile("global_load_async_to_lds_b128 %0, %1, %2"
                         :: "v"(ldsa), "v"(voff), "s"(w_bf)
                         : "memory");
        }
    };

    v8f acc[2][4] = {};

    stageX(0, 0);
    stageW(0, 0);

    for (int kt = 0; kt < KSTEPS; ++kt) {
        const int buf = kt & 1;
        // our async W copies (and ds_stores) must land before the whole
        // workgroup is released to read this buffer
        asm volatile("s_wait_asynccnt 0" ::: "memory");
        __syncthreads();
        if (kt + 1 < KSTEPS) {               // prefetch next tile into buf^1
            stageX(buf ^ 1, (kt + 1) * TILE_K);
            stageW(buf ^ 1, (kt + 1) * TILE_K);
        }

        // --- B fragments (32x16 bf16): lane half selects K 0..15 / 16..31 ---
        v16bf bfrag[4];
        #pragma unroll
        for (int ni = 0; ni < 4; ++ni) {
            const __bf16* bp = &lW[buf][(waveN + ni * 16 + lr) * LDSS + half * 16];
            const v8bf b0 = *(const v8bf*)(bp);
            const v8bf b1 = *(const v8bf*)(bp + 8);
            bfrag[ni] = __builtin_shufflevector(b0, b1,
                0, 1, 2, 3, 4, 5, 6, 7, 8, 9, 10, 11, 12, 13, 14, 15);
        }
        // --- A fragments (16x32 bf16): half selects K 0..15 / 16..31
        #pragma unroll
        for (int mi = 0; mi < 2; ++mi) {
            const __bf16* ap = &lX[buf][(waveM + mi * 16 + lr) * LDSS];
            const v8bf a0 = *(const v8bf*)(ap + half * 16);
            const v8bf a1 = *(const v8bf*)(ap + half * 16 + 8);
            const v16bf afrag = __builtin_shufflevector(a0, a1,
                0, 1, 2, 3, 4, 5, 6, 7, 8, 9, 10, 11, 12, 13, 14, 15);
            #pragma unroll
            for (int ni = 0; ni < 4; ++ni) {
                acc[mi][ni] = __builtin_amdgcn_wmma_f32_16x16x32_bf16(
                    /*neg_a=*/false, afrag,
                    /*neg_b=*/false, bfrag[ni],
                    /*c_mod=*/(short)0, acc[mi][ni],
                    /*reuse_a=*/false, /*reuse_b=*/false);
            }
        }
    }

    // --- epilogue: D VGPR r holds M = r (+8 for lane-half 1), N = lane%16 ---
    #pragma unroll
    for (int mi = 0; mi < 2; ++mi) {
        #pragma unroll
        for (int ni = 0; ni < 4; ++ni) {
            const int col = nTile + waveN + ni * 16 + lr;
            const float bv = bias[col];
            #pragma unroll
            for (int r = 0; r < 8; ++r) {
                const int row = mTile + waveM + mi * 16 + half * 8 + r;
                out[(size_t)row * OUT_DIM + col] = acc[mi][ni][r] + bv;
            }
        }
    }
}

// ---------------------------------------------------------------------------
// launch
// ---------------------------------------------------------------------------
extern "C" void kernel_launch(void* const* d_in, const int* in_sizes, int n_in,
                              void* d_out, int out_size, void* d_ws, size_t ws_size,
                              hipStream_t stream) {
    const float* x        = (const float*)d_in[0];   // [4,2048,4096]
    const int*   q_packed = (const int*)  d_in[1];   // [NPACKED]
    const float* q_scales = (const float*)d_in[2];   // [N_GROUPS]
    const float* q_zeros  = (const float*)d_in[3];   // [N_GROUPS]
    const float* delta    = (const float*)d_in[4];   // [OUT,IN]
    const float* bias     = (const float*)d_in[5];   // [OUT]
    float*       out      = (float*)d_out;           // [4,2048,4096]

    __bf16* w_bf = (__bf16*)d_ws;                    // 32 MB of workspace

    // 1) dequant int4 + delta -> bf16 W_eff
    {
        dim3 grid(NPACKED / 256), block(256);
        figlin_dequant_bf16<<<grid, block, 0, stream>>>(q_packed, q_scales,
                                                        q_zeros, delta, w_bf);
    }
    // 2) WMMA GEMM with fused bias
    {
        dim3 grid(OUT_DIM / TILE_N, M_TOTAL / TILE_M);  // 32 x 64 blocks
        dim3 block(256);
        figlin_wmma_gemm<<<grid, block, 0, stream>>>(x, w_bf, bias, out);
    }
}
